// SMP_2422361555584
// MI455X (gfx1250) — matrix operand order repaired and checked by reading, not compiled, not run on hardware
//
#include <hip/hip_runtime.h>
#include <hip/hip_bf16.h>
#include <math.h>

// ---------------- static problem config (matches reference) ----------------
#define kB   256      // graphs
#define kP   32       // nodes per graph (= n_colors)
#define kN   8192     // total nodes
#define kE   65536    // edges
#define kC   32       // colors
#define kHU  32       // hidden units per (node,color) row
#define kOU  64
#define kHG  128
#define kL   4
#define S2S  19
#define kEPS 1e-5f

typedef __attribute__((ext_vector_type(16))) _Float16 v16h;
typedef __attribute__((ext_vector_type(8)))  _Float16 v8h;
typedef __attribute__((ext_vector_type(4)))  _Float16 v4h;
typedef __attribute__((ext_vector_type(8)))  float    v8f;

__device__ __forceinline__ float sigm(float x) { return 1.0f / (1.0f + expf(-x)); }

// =====================================================================
//  WMMA GEMM, compile-time shape:  C[M,N] = act(A[M,K] @ B + bias)
//  TRANSB=0 -> W stored (K,N) row-major;  TRANSB=1 -> W stored (N,K) (use W^T)
//  M % 128 == 0. Block = 256 thr = 8 waves (wave32); each wave owns a 16x32
//  output tile: two v_wmma_f32_16x16x32_f16 per K-chunk sharing the A frag.
//  Fragment feeds are contiguous ds_load_b128 pairs:
//   A frag (ISA 7.12.2): lane group g needs K in {8g..8g+7} U {16+8g..16+8g+7}
//   B frag: K in {8g..8g+7} U {16+8g..16+8g+7} (same as A) for fixed N -> LDS B panel stored transposed [n][k]
//  ACT: 0=none, 1=relu, 2=leaky_relu(0.01)
// =====================================================================
template<int N, int K, int TRANSB, int ACT>
__global__ void __launch_bounds__(256)
k_gemm(const float* __restrict__ A, const float* __restrict__ W,
       const float* __restrict__ bias, float* __restrict__ C, int M)
{
    __shared__ __align__(16) _Float16 sA[128 * 32];   // 8 KB
    __shared__ __align__(16) _Float16 sBt[32 * K];    // <= 16 KB, [n][k]

    const int tid  = threadIdx.x;
    const int wave = tid >> 5;
    const int lane = tid & 31;
    const int grp  = lane >> 4;          // wave32 lane group
    const int lan  = lane & 15;
    const int rowBase = blockIdx.x * 128;
    const int n0      = blockIdx.y * 32;

    // stage B panel transposed: sBt[n*K + k]
#pragma unroll
    for (int idx = tid; idx < 32 * K; idx += 256) {
        int n = idx / K, k = idx % K;                 // K pow2 -> shifts
        float w = TRANSB ? W[(size_t)(n0 + n) * K + k]
                         : W[(size_t)k * N + (n0 + n)];
        sBt[idx] = (_Float16)w;
    }
    __syncthreads();

    v8f acc0 = {}, acc1 = {};
    const int arow = (wave * 16 + lan) * 32;

#pragma unroll
    for (int kb = 0; kb < K; kb += 32) {
        // stage A chunk 128x32 as f16: float4 global loads, 8B LDS stores
#pragma unroll
        for (int i = tid; i < 1024; i += 256) {
            int r = i >> 3, kq = (i & 7) * 4;
            float4 f = *(const float4*)&A[(size_t)(rowBase + r) * K + kb + kq];
            v4h h = { (_Float16)f.x, (_Float16)f.y, (_Float16)f.z, (_Float16)f.w };
            *(v4h*)&sA[r * 32 + kq] = h;
        }
        if (kb + 32 < K) {
            // global_prefetch_b8 of next A chunk (probe-confirmed lowering)
            __builtin_prefetch(&A[(size_t)(rowBase + (tid >> 1)) * K + kb + 32], 0, 1);
        }
        __syncthreads();

        v8h al = *(const v8h*)&sA[arow + grp * 8];
        v8h ah = *(const v8h*)&sA[arow + 16 + grp * 8];
        v16h af = __builtin_shufflevector(al, ah, 0,1,2,3,4,5,6,7,8,9,10,11,12,13,14,15);

        v8h b0l = *(const v8h*)&sBt[lan * K + kb + grp * 8];
        v8h b0h = *(const v8h*)&sBt[lan * K + kb + 16 + grp * 8];
        v16h bf0 = __builtin_shufflevector(b0l, b0h, 0,1,2,3,4,5,6,7,8,9,10,11,12,13,14,15);

        v8h b1l = *(const v8h*)&sBt[(lan + 16) * K + kb + grp * 8];
        v8h b1h = *(const v8h*)&sBt[(lan + 16) * K + kb + 16 + grp * 8];
        v16h bf1 = __builtin_shufflevector(b1l, b1h, 0,1,2,3,4,5,6,7,8,9,10,11,12,13,14,15);

        acc0 = __builtin_amdgcn_wmma_f32_16x16x32_f16(false, af, false, bf0, (short)0, acc0, false, false);
        acc1 = __builtin_amdgcn_wmma_f32_16x16x32_f16(false, af, false, bf1, (short)0, acc1, false, false);
        __syncthreads();
    }

    // C/D layout: VGPR r -> M = r + 8*grp, N = lane&15
    float bv0 = bias[n0 + lan];
    float bv1 = bias[n0 + 16 + lan];
#pragma unroll
    for (int r = 0; r < 8; ++r) {
        int m = rowBase + wave * 16 + grp * 8 + r;
        float v0 = acc0[r] + bv0, v1 = acc1[r] + bv1;
        if (ACT == 1) { v0 = fmaxf(v0, 0.0f); v1 = fmaxf(v1, 0.0f); }
        if (ACT == 2) { v0 = (v0 > 0.f) ? v0 : 0.01f * v0;
                        v1 = (v1 > 0.f) ? v1 : 0.01f * v1; }
        C[(size_t)m * N + n0 + lan]      = v0;
        C[(size_t)m * N + n0 + 16 + lan] = v1;
    }
}

// host-side dispatcher over the shapes this model uses
static void gemm(const float* A, const float* W, const float* bias, float* C,
                 int M, int N, int K, int transB, int act, hipStream_t s) {
    dim3 grid(M / 128, N / 32);
#define GEMM_CASE(NN_, KK_, TB_, AC_)                                          \
    if (N == NN_ && K == KK_ && transB == TB_ && act == AC_) {                 \
        k_gemm<NN_, KK_, TB_, AC_><<<grid, 256, 0, s>>>(A, W, bias, C, M);     \
        return;                                                                \
    }
    GEMM_CASE(32, 32, 0, 0)      // u2u block-diagonal terms
    GEMM_CASE(64, 64, 0, 1)      // extractor MLP
    GEMM_CASE(384, 64, 1, 0)     // GRU input gates
    GEMM_CASE(384, 128, 1, 0)    // GRU hidden gates
    GEMM_CASE(128, 128, 0, 2)    // node MLP (leaky)
    GEMM_CASE(128, 256, 0, 1)    // graph head layer 0 (relu)
#undef GEMM_CASE
}

// ---------------------------------------------------------------------
//  small utility kernels
// ---------------------------------------------------------------------
__global__ void k_zero_f(float* p, size_t n) {
    size_t t = (size_t)blockIdx.x * 256 + threadIdx.x;
    if (t < n) p[t] = 0.0f;
}
__global__ void k_zero_i(int* p, int n) {
    int t = blockIdx.x * 256 + threadIdx.x;
    if (t < n) p[t] = 0;
}

// ---- edge binning --------------------------------------------------
__global__ void k_count_edges(const int* __restrict__ ei, int* cnt) {
    int e = blockIdx.x * 256 + threadIdx.x;
    if (e < kE) atomicAdd(&cnt[ei[e] >> 5], 1);        // g = src / P
}
__global__ void k_scan_edges(const int* __restrict__ cnt, int* offs, int* cursor,
                             float* avg_e) {
    if (threadIdx.x == 0) {
        int run = 0;
        for (int g = 0; g < kB; ++g) { offs[g] = run; cursor[g] = run; run += cnt[g]; }
    }
    int g = threadIdx.x;
    if (g < kB) avg_e[g] = (float)cnt[g] / (float)kP;
    __syncthreads();
}
__global__ void k_bin_edges(const int* __restrict__ ei, int* cursor, int* ebin) {
    int e = blockIdx.x * 256 + threadIdx.x;
    if (e < kE) {
        int g = ei[e] >> 5;
        int pos = atomicAdd(&cursor[g], 1);
        ebin[pos] = e;
    }
}

// ---- input embedding ----------------------------------------------
__global__ void k_gc(const float* __restrict__ x, const int* __restrict__ coloring,
                     float* gc) {
    int n = blockIdx.x * 256 + threadIdx.x;
    if (n < kN && x[n * 2 + 0] != 0.0f) gc[(n >> 5) * kC + coloring[n]] = 1.0f;
}
__global__ void k_build_u0(const float* __restrict__ x, const int* __restrict__ coloring,
                           const float* __restrict__ gc, const float* __restrict__ lw,
                           const float* __restrict__ lb, float* __restrict__ u) {
    size_t t = (size_t)blockIdx.x * 256 + threadIdx.x;
    if (t >= (size_t)kN * kC * kHU) return;
    int h = t & 31, c = (t >> 5) & 31, n = (int)(t >> 10);
    int g = n >> 5;
    float onehot = (c == coloring[n]) ? 1.0f : 0.0f;
    float usp = gc[g * kC + c];
    float ulap = onehot * x[n * 2 + 1];
    u[t] = onehot * lw[0 * kHU + h] + usp * lw[1 * kHU + h] + ulap * lw[2 * kHU + h] + lb[h];
}

// ---- training-mode batch norm over (N*C, 32) -----------------------
__global__ void k_bn_reduce(const float* __restrict__ u, float* stats, size_t total) {
    int tid = threadIdx.x;
    float s = 0.f, s2 = 0.f;
    for (size_t i = (size_t)blockIdx.x * 256 + tid; i < total; i += (size_t)gridDim.x * 256) {
        float v = u[i]; s += v; s2 += v * v;          // channel = i&31 == tid&31
    }
    __shared__ float ls[256], ls2[256];
    ls[tid] = s; ls2[tid] = s2; __syncthreads();
    if (tid < 32) {
        float a = 0.f, b = 0.f;
        for (int j = tid; j < 256; j += 32) { a += ls[j]; b += ls2[j]; }
        atomicAdd(&stats[tid], a); atomicAdd(&stats[32 + tid], b);
    }
}
__global__ void k_bn_apply(float* u, const float* __restrict__ stats, size_t total) {
    size_t t = (size_t)blockIdx.x * 256 + threadIdx.x;
    if (t >= total) return;
    int h = t & 31;
    float cntf = (float)(total >> 5);
    float m = stats[h] / cntf;
    float var = stats[32 + h] / cntf - m * m;
    u[t] = (u[t] - m) * rsqrtf(var + kEPS);
}

// ---- u2u helpers ---------------------------------------------------
// expand 4 grouped (G=8, 4x4) weights into 4 dense 32x32 block-diagonal mats
__global__ void k_expand_bd(const float* w0, const float* w1, const float* w2,
                            const float* w3, float* out) {
    int t = blockIdx.x * 256 + threadIdx.x;
    if (t >= 4096) return;
    int mat = t >> 10, idx = t & 1023;
    int k = idx >> 5, n = idx & 31;
    const float* W = (mat == 0) ? w0 : (mat == 1) ? w1 : (mat == 2) ? w2 : w3;
    out[t] = ((k >> 2) == (n >> 2)) ? W[(k >> 2) * 16 + (k & 3) * 4 + (n & 3)] : 0.0f;
}
// color-mean over C: out[n*ldo + off + h]
__global__ void k_meanC(const float* __restrict__ u, float* out, int ldo, int off) {
    int t = blockIdx.x * 256 + threadIdx.x;      // N*32
    if (t >= kN * kHU) return;
    int h = t & 31, n = t >> 5;
    float s = 0.f;
    for (int c = 0; c < kC; ++c) s += u[(size_t)n * 1024 + c * 32 + h];
    out[(size_t)n * ldo + off + h] = s * (1.0f / kC);
}
// own-color row gather: out[n*ldo + off + h]
__global__ void k_own(const float* __restrict__ u, const int* __restrict__ coloring,
                      float* out, int ldo, int off) {
    int t = blockIdx.x * 256 + threadIdx.x;
    if (t >= kN * kHU) return;
    int h = t & 31, n = t >> 5;
    out[(size_t)n * ldo + off + h] = u[(size_t)n * 1024 + coloring[n] * 32 + h];
}
// graph mean: gm[(b*32+c)*32+h] = mean_p u[(b*32+p), c, h]
__global__ void k_gm(const float* __restrict__ u, float* gm) {
    int t = blockIdx.x * 256 + threadIdx.x;      // B*C*32
    if (t >= kB * kC * kHU) return;
    int h = t & 31, c = (t >> 5) & 31, b = t >> 10;
    float s = 0.f;
    for (int p = 0; p < kP; ++p) s += u[(size_t)(b * kP + p) * 1024 + c * 32 + h];
    gm[t] = s * (1.0f / kP);
}
// u2 += t2[n] + t3[n] + t4[batch,c]   (t1 already in u2 from GEMM)
__global__ void k_combine_u2(float* u2, const float* __restrict__ t2,
                             const float* __restrict__ t3, const float* __restrict__ t4) {
    size_t t = (size_t)blockIdx.x * 256 + threadIdx.x;
    if (t >= (size_t)kN * kC * kHU) return;
    int h = t & 31, c = (int)((t >> 5) & 31), n = (int)(t >> 10);
    int g = n >> 5;
    u2[t] += t2[(size_t)n * 32 + h] + t3[(size_t)n * 32 + h]
           + t4[(size_t)(g * kC + c) * 32 + h];
}

// ---- conflict-free per-graph edge aggregation ---------------------
// one block per graph; thread owns elements {tid, tid+256, tid+512, tid+768}
// of every node's 1024-float message -> no atomics, no barriers needed.
__global__ void __launch_bounds__(256)
k_aggr(const float* __restrict__ u2, const int* __restrict__ ei,
       const int* __restrict__ ebin, const int* __restrict__ offs,
       const int* __restrict__ cnt, float* __restrict__ aggr) {
    int g = blockIdx.x, tid = threadIdx.x;
    size_t base = (size_t)g * kP * 1024;
    for (int nd = 0; nd < kP; ++nd)
#pragma unroll
        for (int e4 = 0; e4 < 4; ++e4)
            aggr[base + nd * 1024 + e4 * 256 + tid] = 0.0f;
    int beg = offs[g], num = cnt[g];
    for (int i = 0; i < num; ++i) {
        int e = ebin[beg + i];
        int sl = ei[e] & 31, dl = ei[kE + e] & 31;
#pragma unroll
        for (int e4 = 0; e4 < 4; ++e4) {
            int el = e4 * 256 + tid;
            aggr[base + dl * 1024 + el] += u2[base + sl * 1024 + el];
        }
    }
}
// out = (u2 + aggr + alpha*u2*aggr) / avg_e[g]
__global__ void k_smp_out(const float* __restrict__ u2, const float* __restrict__ aggr,
                          const float* __restrict__ alpha, const float* __restrict__ avg_e,
                          float* __restrict__ u) {
    size_t t = (size_t)blockIdx.x * 256 + threadIdx.x;
    if (t >= (size_t)kN * kC * kHU) return;
    int h = t & 31, g = (int)(t >> 15);          // t / (32*32*32) = node/32
    float a = u2[t], b = aggr[t];
    u[t] = (a + b + alpha[h] * a * b) / avg_e[g];
}

// ---- GRU gate combine ---------------------------------------------
__global__ void k_gru(const float* __restrict__ gi, const float* __restrict__ gh,
                      float* __restrict__ h) {
    size_t t = (size_t)blockIdx.x * 256 + threadIdx.x;
    if (t >= (size_t)kN * kHG) return;
    int j = t & 127; size_t n = t >> 7;
    size_t b = n * 384;
    float r  = sigm(gi[b + j]       + gh[b + j]);
    float z  = sigm(gi[b + 128 + j] + gh[b + 128 + j]);
    float nt = tanhf(gi[b + 256 + j] + r * gh[b + 256 + j]);
    h[t] = (1.0f - z) * nt + z * h[t];
}

// ---- node head (last 128->2 layer) --------------------------------
__global__ void k_node_out(const float* __restrict__ v, const float* __restrict__ w,
                           const float* __restrict__ b, float* __restrict__ out) {
    int t = blockIdx.x * 256 + threadIdx.x;      // N*2
    if (t >= kN * 2) return;
    int o = t & 1, n = t >> 1;
    float s = b[o];
    for (int k = 0; k < kHG; ++k) s += v[(size_t)n * kHG + k] * w[k * 2 + o];
    out[t] = s;
}

// ---- fused Set2Set: one block per graph, all 19 steps in LDS -------
// xg staging uses GLOBAL_LOAD_ASYNC_TO_LDS_B128 (ASYNCcnt) -- a verbatim
// global->LDS byte copy, drained with s_wait_asynccnt before first use.
__global__ void __launch_bounds__(128)
k_set2set(const float* __restrict__ h, const float* __restrict__ wih,
          const float* __restrict__ whh, const float* __restrict__ bih,
          const float* __restrict__ bhh, float* __restrict__ q) {
    __shared__ __align__(16) float xg[kP * kHG];               // 16 KB
    __shared__ float qs[256], hS[128], cS[128], gates[512], att[32];
    int g = blockIdx.x, tid = threadIdx.x;
    // async copy 32x128 f32 tile: 1024 x b128, 8 per lane
    for (int i = tid; i < 1024; i += 128) {
        const float* gp = h + (size_t)g * kP * kHG + i * 4;
        unsigned la = (unsigned)(uintptr_t)&xg[i * 4];   // low 32b = LDS offset
        asm volatile("global_load_async_to_lds_b128 %0, %1, off"
                     :: "v"(la), "v"(gp) : "memory");
    }
    hS[tid] = 0.f; cS[tid] = 0.f; qs[tid] = 0.f; qs[128 + tid] = 0.f;
    asm volatile("s_wait_asynccnt 0" ::: "memory");
    __syncthreads();
    for (int step = 0; step < S2S; ++step) {
        for (int jj = 0; jj < 4; ++jj) {
            int j = jj * 128 + tid;
            float acc = bih[j] + bhh[j];
            for (int k = 0; k < 256; ++k) acc += qs[k] * wih[(size_t)j * 256 + k];
            for (int k = 0; k < 128; ++k) acc += hS[k] * whh[(size_t)j * 128 + k];
            gates[j] = acc;
        }
        __syncthreads();
        float c2 = sigm(gates[128 + tid]) * cS[tid] + sigm(gates[tid]) * tanhf(gates[256 + tid]);
        float h2 = sigm(gates[384 + tid]) * tanhf(c2);
        __syncthreads();
        cS[tid] = c2; hS[tid] = h2;
        __syncthreads();
        if (tid < 32) { float a = 0.f; for (int k = 0; k < 128; ++k) a += xg[tid * 128 + k] * hS[k]; att[tid] = a; }
        __syncthreads();
        if (tid == 0) {
            float mx = att[0]; for (int p = 1; p < 32; ++p) mx = fmaxf(mx, att[p]);
            float s = 0.f; for (int p = 0; p < 32; ++p) { att[p] = expf(att[p] - mx); s += att[p]; }
            float inv = 1.0f / s; for (int p = 0; p < 32; ++p) att[p] *= inv;
        }
        __syncthreads();
        float r = 0.f;
        for (int p = 0; p < 32; ++p) r += att[p] * xg[p * 128 + tid];
        qs[tid] = hS[tid]; qs[128 + tid] = r;
        __syncthreads();
    }
    q[(size_t)g * 256 + tid] = qs[tid];
    q[(size_t)g * 256 + 128 + tid] = qs[128 + tid];
}

// ---- small BN for graph head (rows x 128, one block) ---------------
__global__ void k_bn_small(float* v, int rows) {
    int c = threadIdx.x;                          // 128 channels
    float s = 0.f, s2 = 0.f;
    for (int r = 0; r < rows; ++r) { float x = v[(size_t)r * 128 + c]; s += x; s2 += x * x; }
    float m = s / rows, var = s2 / rows - m * m;
    float inv = rsqrtf(var + kEPS);
    for (int r = 0; r < rows; ++r) v[(size_t)r * 128 + c] = (v[(size_t)r * 128 + c] - m) * inv;
}
__global__ void k_graph_out(const float* __restrict__ g2, const float* __restrict__ w,
                            const float* __restrict__ b, float* __restrict__ out) {
    int t = blockIdx.x * 256 + threadIdx.x;       // B*2
    if (t >= kB * 2) return;
    int o = t & 1, r = t >> 1;
    float s = b[o];
    for (int k = 0; k < kHG; ++k) s += g2[(size_t)r * kHG + k] * w[k * 2 + o];
    out[t] = s;
}

// =====================================================================
//  orchestration
// =====================================================================
static inline float* alloc_f(char*& p, size_t n) {
    float* r = (float*)p; p += ((n * sizeof(float) + 255) / 256) * 256; return r;
}
static inline int* alloc_i(char*& p, size_t n) {
    int* r = (int*)p; p += ((n * sizeof(int) + 255) / 256) * 256; return r;
}
static inline dim3 g1d(size_t n) { return dim3((unsigned)((n + 255) / 256)); }

extern "C" void kernel_launch(void* const* d_in, const int* in_sizes, int n_in,
                              void* d_out, int out_size, void* d_ws, size_t ws_size,
                              hipStream_t stream) {
    // ---- inputs (dict-insertion-order flattening) ----
    const float* x        = (const float*)d_in[0];
    const int*   ei       = (const int*)d_in[1];     // (2,E) flat
    const int*   coloring = (const int*)d_in[3];
    auto F = [&](int i) { return (const float*)d_in[i]; };
    const int LIN_W = 4, LIN_B = 5;
    auto LW  = [&](int l, int j) { return F(6 + 9 * l + j); };       // u2u_w[j]
    auto LBv = [&](int l, int j) { return F(6 + 9 * l + 4 + j); };   // u2u_b[j]
    auto LA  = [&](int l) { return F(6 + 9 * l + 8); };              // alpha
    const int EXT_W1 = 42, EXT_B1 = 43, EXT_W2 = 44, EXT_B2 = 45;
    const int GRU_WIH = 46, GRU_WHH = 47, GRU_BIH = 48, GRU_BHH = 49;
    const int LSTM_WIH = 50, LSTM_WHH = 51, LSTM_BIH = 52, LSTM_BHH = 53;
    const int NM = 54;   // node_mlp: (w,b)x3
    const int GM = 60;   // graph_mlp: (w,b)x3
    float* out = (float*)d_out;

    // ---- workspace carve-out ----
    char* wp = (char*)d_ws;
    float* u     = alloc_f(wp, (size_t)kN * kC * kHU);
    float* u2    = alloc_f(wp, (size_t)kN * kC * kHU);
    float* aggr  = alloc_f(wp, (size_t)kN * kC * kHU);
    float* t2    = alloc_f(wp, (size_t)kN * 32);
    float* t3    = alloc_f(wp, (size_t)kN * 32);
    float* t4    = alloc_f(wp, (size_t)kB * kC * 32);
    float* umean = alloc_f(wp, (size_t)kN * 32);
    float* own   = alloc_f(wp, (size_t)kN * 32);
    float* gmb   = alloc_f(wp, (size_t)kB * kC * 32);
    float* z     = alloc_f(wp, (size_t)kN * 64);
    float* e1    = alloc_f(wp, (size_t)kN * 64);
    float* extb  = alloc_f(wp, (size_t)kN * 64);
    float* gi    = alloc_f(wp, (size_t)kN * 384);
    float* gh    = alloc_f(wp, (size_t)kN * 384);
    float* hbuf  = alloc_f(wp, (size_t)kN * kHG);
    float* v1    = alloc_f(wp, (size_t)kN * kHG);
    float* v2    = alloc_f(wp, (size_t)kN * kHG);
    float* Wbd   = alloc_f(wp, 4 * 32 * 32);
    float* gc    = alloc_f(wp, (size_t)kB * kC);
    float* avg_e = alloc_f(wp, kB);
    float* stats = alloc_f(wp, 64);
    float* q     = alloc_f(wp, (size_t)kB * 256);
    float* g1b   = alloc_f(wp, (size_t)kB * kHG);
    float* g2b   = alloc_f(wp, (size_t)kB * kHG);
    int* cnt    = alloc_i(wp, kB);
    int* offs   = alloc_i(wp, kB);
    int* cursor = alloc_i(wp, kB);
    int* ebin   = alloc_i(wp, kE);

    size_t TOT = (size_t)kN * kC * kHU;

    // ---- edge binning + avg_edges ----
    k_zero_i<<<g1d(kB), 256, 0, stream>>>(cnt, kB);
    k_count_edges<<<g1d(kE), 256, 0, stream>>>(ei, cnt);
    k_scan_edges<<<1, 256, 0, stream>>>(cnt, offs, cursor, avg_e);
    k_bin_edges<<<g1d(kE), 256, 0, stream>>>(ei, cursor, ebin);

    // ---- input embedding ----
    k_zero_f<<<g1d(kB * kC), 256, 0, stream>>>(gc, (size_t)kB * kC);
    k_gc<<<g1d(kN), 256, 0, stream>>>(x, coloring, gc);
    k_build_u0<<<g1d(TOT), 256, 0, stream>>>(x, coloring, gc, F(LIN_W), F(LIN_B), u);
    k_zero_f<<<g1d((size_t)kN * kHG), 256, 0, stream>>>(hbuf, (size_t)kN * kHG);

    for (int l = 0; l < kL; ++l) {
        if (l > 0) {
            k_zero_f<<<1, 256, 0, stream>>>(stats, 64);
            k_bn_reduce<<<256, 256, 0, stream>>>(u, stats, TOT);
            k_bn_apply<<<g1d(TOT), 256, 0, stream>>>(u, stats, TOT);
        }
        // ---- _u2u via WMMA on block-diagonal 32x32 weights ----
        k_expand_bd<<<16, 256, 0, stream>>>(LW(l, 0), LW(l, 1), LW(l, 2), LW(l, 3), Wbd);
        k_meanC<<<g1d(kN * 32), 256, 0, stream>>>(u, umean, 32, 0);
        k_own<<<g1d(kN * 32), 256, 0, stream>>>(u, coloring, own, 32, 0);
        k_gm<<<g1d(kB * kC * 32), 256, 0, stream>>>(u, gmb);
        gemm(u,     Wbd + 0,    LBv(l, 0), u2, kN * kC, 32, 32, 0, 0, stream);
        gemm(umean, Wbd + 1024, LBv(l, 1), t2, kN,      32, 32, 0, 0, stream);
        gemm(own,   Wbd + 2048, LBv(l, 2), t3, kN,      32, 32, 0, 0, stream);
        gemm(gmb,   Wbd + 3072, LBv(l, 3), t4, kB * kC, 32, 32, 0, 0, stream);
        k_combine_u2<<<g1d(TOT), 256, 0, stream>>>(u2, t2, t3, t4);
        // ---- message passing + SMP combine ----
        k_aggr<<<kB, 256, 0, stream>>>(u2, ei, ebin, offs, cnt, aggr);
        k_smp_out<<<g1d(TOT), 256, 0, stream>>>(u2, aggr, LA(l), avg_e, u);
        // ---- NodeExtractor: z = [own | meanC]; two relu GEMMs ----
        k_own<<<g1d(kN * 32), 256, 0, stream>>>(u, coloring, z, 64, 0);
        k_meanC<<<g1d(kN * 32), 256, 0, stream>>>(u, z, 64, 32);
        gemm(z,  F(EXT_W1), F(EXT_B1), e1,   kN, 64, 64, 0, 1, stream);
        gemm(e1, F(EXT_W2), F(EXT_B2), extb, kN, 64, 64, 0, 1, stream);
        // ---- GRU: gi = ext @ Wih^T, gh = h @ Whh^T ----
        gemm(extb, F(GRU_WIH), F(GRU_BIH), gi, kN, 384, 64,  1, 0, stream);
        gemm(hbuf, F(GRU_WHH), F(GRU_BHH), gh, kN, 384, 128, 1, 0, stream);
        k_gru<<<g1d((size_t)kN * kHG), 256, 0, stream>>>(gi, gh, hbuf);
    }

    // ---- node head ----
    gemm(hbuf, F(NM + 0), F(NM + 1), v1, kN, 128, 128, 0, 2, stream);
    gemm(v1,   F(NM + 2), F(NM + 3), v2, kN, 128, 128, 0, 2, stream);
    k_node_out<<<g1d(kN * 2), 256, 0, stream>>>(v2, F(NM + 4), F(NM + 5), out);

    // ---- Set2Set + graph head ----
    k_set2set<<<kB, 128, 0, stream>>>(hbuf, F(LSTM_WIH), F(LSTM_WHH), F(LSTM_BIH), F(LSTM_BHH), q);
    gemm(q,   F(GM + 0), F(GM + 1), g1b, kB, 128, 256, 0, 1, stream);
    k_bn_small<<<1, 128, 0, stream>>>(g1b, kB);
    gemm(g1b, F(GM + 2), F(GM + 3), g2b, kB, 128, 128, 0, 2, stream);
    k_bn_small<<<1, 128, 0, stream>>>(g2b, kB);
    k_graph_out<<<g1d(kB * 2), 256, 0, stream>>>(g2b, F(GM + 4), F(GM + 5), out + kN * 2);
}
